// RNNHead_77060303225371
// MI455X (gfx1250) — matrix operand order repaired and checked by reading, not compiled, not run on hardware
//
#include <hip/hip_runtime.h>

// GRU dims (compile-time)
#define BB   16
#define TT   4096
#define DIMX 256
#define HH   128
#define H3   384          // 3*H
#define BTN  (BB * TT)    // 65536 rows

typedef __attribute__((ext_vector_type(16))) __bf16        v16bf;
typedef __attribute__((ext_vector_type(8)))  float         v8f;
typedef __attribute__((ext_vector_type(4)))  unsigned int  u32x4;

union BF16Frag {
    unsigned int u[8];
    u32x4        q[2];
    v16bf        v;
};

// Fragment geometry: one 16(N) x 32(K) bf16 B-fragment = 32 lanes x 16 elems
#define FRAG_ELEMS   512                     // 32 * 16 bf16
#define WI_FRAGS     192                     // (256/32) kchunks * 24 N-tiles
#define WH_FRAGS     96                      // (128/32) kchunks * 24 N-tiles

// ---------------------------------------------------------------------------
// f32 -> bf16 conversion (layer-0 input staging)
// ---------------------------------------------------------------------------
__global__ void cvt_f32_bf16_kernel(const float* __restrict__ src,
                                    __bf16* __restrict__ dst, int n) {
    int i = blockIdx.x * blockDim.x + threadIdx.x;
    if (i < n) dst[i] = (__bf16)src[i];
}

// ---------------------------------------------------------------------------
// One-shot weight prep: [slice, K, 384] f32  ->  bf16 WMMA B-fragment layout.
// Fragment f of slice s covers N-tile (f / (K/32)), k-chunk (f % (K/32)).
// Lane (n,half) element j corresponds to W[kc*32 + half*16 + j][tile*16 + n].
// grid = (ceil(frags/8), slices), block = 256 (8 waves, 1 frag/wave).
// ---------------------------------------------------------------------------
__global__ __launch_bounds__(256) void prep_weights_kernel(
        const float* __restrict__ src,
        __bf16*      __restrict__ dst,
        int K) {
    const int lane = threadIdx.x & 31;
    const int wave = threadIdx.x >> 5;
    const int kcCount = K >> 5;
    const int frags = 24 * kcCount;
    const int f = blockIdx.x * 8 + wave;
    if (f >= frags) return;
    const int s    = blockIdx.y;
    const int tile = f / kcCount;
    const int kc   = f - tile * kcCount;
    const int n    = lane & 15;
    const int half = lane >> 4;

    const float* w = src + (size_t)s * K * H3
                         + (size_t)(kc * 32 + half * 16) * H3 + tile * 16 + n;
    BF16Frag b;
#pragma unroll
    for (int j = 0; j < 16; ++j) b.v[j] = (__bf16)w[(size_t)j * H3];

    __bf16* o = dst + ((size_t)(s * frags + f) * 32 + lane) * 16;
    *(u32x4*)o       = b.q[0];
    *((u32x4*)o + 1) = b.q[1];
}

// ---------------------------------------------------------------------------
// xg = act @ Wi  for one layer, both directions (prepped bf16 weights).
//   act  : [BT, 256] bf16 row-major
//   WiPL : [2, WI_FRAGS, 512] bf16 fragments (layer slice)
//   xg   : [2, BT, 384] bf16
// grid = (BT/128, 384/64, 2), block = 256 (8 waves).
// Inner loop: 2x b128 A loads + per tile 2x b128 B loads + v_wmma. No cvt.
// ---------------------------------------------------------------------------
__global__ __launch_bounds__(256) void xg_gemm_kernel(
        const __bf16* __restrict__ act,
        const __bf16* __restrict__ WiPL,
        __bf16*       __restrict__ xg) {
    const int lane = threadIdx.x & 31;
    const int wave = threadIdx.x >> 5;
    const int nm   = lane & 15;
    const int half = lane >> 4;
    const int dir  = blockIdx.z;

    const int rowBase  = blockIdx.x * 128 + wave * 16;
    const int tileBase = blockIdx.y * 4;          // 4 N-tiles of 16

    const __bf16* WiPd = WiPL + (size_t)dir * WI_FRAGS * FRAG_ELEMS;
    __bf16*       xgd  = xg   + (size_t)dir * BTN * H3;

    const v8f vzero = {0.f, 0.f, 0.f, 0.f, 0.f, 0.f, 0.f, 0.f};
    v8f acc[4];
#pragma unroll
    for (int i = 0; i < 4; ++i) acc[i] = vzero;

    for (int kc = 0; kc < 8; ++kc) {            // K = 256 in chunks of 32
        // A fragment: two contiguous 16B loads per lane
        BF16Frag a;
        const char* ab = (const char*)(act + (size_t)(rowBase + nm) * DIMX + kc * 32)
                       + (half << 5);
        a.q[0] = *(const u32x4*)ab;
        a.q[1] = *(const u32x4*)(ab + 16);
#pragma unroll
        for (int tt = 0; tt < 4; ++tt) {
            BF16Frag b;
            const __bf16* bp = WiPd
                + ((size_t)((tileBase + tt) * 8 + kc) * 32 + lane) * 16;
            b.q[0] = ((const u32x4*)bp)[0];
            b.q[1] = ((const u32x4*)bp)[1];
            acc[tt] = __builtin_amdgcn_wmma_f32_16x16x32_bf16(
                false, a.v, false, b.v, (short)0, acc[tt], false, false);
        }
    }
    // D layout: VGPR v -> M = v + 8*half, N = nm
#pragma unroll
    for (int tt = 0; tt < 4; ++tt) {
#pragma unroll
        for (int v = 0; v < 8; ++v) {
            const int row = rowBase + v + (half << 3);
            xgd[(size_t)row * H3 + (tileBase + tt) * 16 + nm] = (__bf16)acc[tt][v];
        }
    }
}

// ---------------------------------------------------------------------------
// GRU scan over time for one layer, one direction per block.
// grid = 2 (dir), block = 384 (12 waves x 2 tiles = 24 N-tiles).
// Wh bf16 B-fragments held in VGPRs for all 4096 steps (64 VGPRs/wave).
// ---------------------------------------------------------------------------
__global__ __launch_bounds__(384) void gru_scan_kernel(
        const __bf16* __restrict__ xg,
        const __bf16* __restrict__ WhPL,     // prepped fragments, layer slice
        const float*  __restrict__ bhL,
        __bf16*       __restrict__ actOut,   // bf16 out (layers 0,1)
        float*        __restrict__ outF,     // f32 out (last layer)
        int isLast) {
    const int dir  = blockIdx.x;
    const int tid  = threadIdx.x;
    const int lane = tid & 31;
    const int wave = tid >> 5;
    const int nm   = lane & 15;
    const int half = lane >> 4;

    const __bf16* WhPd = WhPL + (size_t)dir * WH_FRAGS * FRAG_ELEMS;
    const float*  bh   = bhL  + (size_t)dir * H3;
    const __bf16* xgd  = xg   + (size_t)dir * BTN * H3;

    __shared__ float  Hc[16 * H3];     // h @ Wh staging  (24 KB)
    __shared__ float  hF[16 * HH];     // h, f32          (8 KB)
    __shared__ __bf16 hB[16 * HH];     // h, bf16 mirror  (4 KB)
    __shared__ float  biasS[H3];

    for (int i = tid; i < 16 * HH; i += 384) { hF[i] = 0.f; hB[i] = (__bf16)0.f; }
    for (int i = tid; i < H3; i += 384) biasS[i] = bh[i];

    // Preload Wh B-fragments: wave owns N-tiles 2w and 2w+1, K=128 in 4 chunks.
    BF16Frag whf[2][4];
#pragma unroll
    for (int i = 0; i < 2; ++i) {
        const int tile = wave * 2 + i;
#pragma unroll
        for (int kt = 0; kt < 4; ++kt) {
            const __bf16* bp = WhPd + ((size_t)(tile * 4 + kt) * 32 + lane) * 16;
            whf[i][kt].q[0] = ((const u32x4*)bp)[0];
            whf[i][kt].q[1] = ((const u32x4*)bp)[1];
        }
    }
    __syncthreads();

    const v8f vzero = {0.f, 0.f, 0.f, 0.f, 0.f, 0.f, 0.f, 0.f};

    for (int step = 0; step < TT; ++step) {
        const int t = dir ? (TT - 1 - step) : step;

        // A fragments of h from LDS: two 16B ds loads per chunk
        BF16Frag ah[4];
#pragma unroll
        for (int kt = 0; kt < 4; ++kt) {
            const char* hb8 = (const char*)(hB + nm * HH + kt * 32) + (half << 5);
            ah[kt].q[0] = *(const u32x4*)hb8;
            ah[kt].q[1] = *(const u32x4*)(hb8 + 16);
        }
        // Hc tiles: 2 per wave, K accumulated over 4 WMMAs each
#pragma unroll
        for (int i = 0; i < 2; ++i) {
            v8f acc = vzero;
#pragma unroll
            for (int kt = 0; kt < 4; ++kt)
                acc = __builtin_amdgcn_wmma_f32_16x16x32_bf16(
                    false, ah[kt].v, false, whf[i][kt].v, (short)0, acc,
                    false, false);
            const int col = (wave * 2 + i) * 16 + nm;
#pragma unroll
            for (int v = 0; v < 8; ++v)
                Hc[(v + (half << 3)) * H3 + col] = acc[v];
        }
        __syncthreads();

        // Gate math: 16x128 h-update elements across 384 threads
        for (int idx = tid; idx < 16 * HH; idx += 384) {
            const int b = idx >> 7;
            const int c = idx & 127;
            const size_t rbase = ((size_t)b * TT + t) * H3;
            const float xr = (float)xgd[rbase + c];
            const float xz = (float)xgd[rbase + HH + c];
            const float xn = (float)xgd[rbase + 2 * HH + c];
            const int hb = b * H3;
            const float r  = 1.f / (1.f + __expf(-(xr + Hc[hb + c] + biasS[c])));
            const float z  = 1.f / (1.f + __expf(-(xz + Hc[hb + HH + c] + biasS[HH + c])));
            const float nn = tanhf(xn + r * (Hc[hb + 2 * HH + c] + biasS[2 * HH + c]));
            const float hnew = (1.f - z) * nn + z * hF[idx];
            hF[idx] = hnew;
            hB[idx] = (__bf16)hnew;
            const size_t obase = ((size_t)b * TT + t) * DIMX + dir * HH + c;
            if (isLast) outF[obase] = hnew;
            else        actOut[obase] = (__bf16)hnew;
            // light prefetch of next timestep's xg row (one line per 64 cols)
            if ((c & 63) == 0) {
                const int tn = dir ? (t - 1) : (t + 1);
                if ((unsigned)tn < (unsigned)TT)
                    __builtin_prefetch(
                        (const void*)&xgd[((size_t)b * TT + tn) * H3 + c], 0, 1);
            }
        }
        __syncthreads();
    }
}

// ---------------------------------------------------------------------------
// Host launch.
// ws layout (bytes):
//   [xg bf16: 2*BT*384*2 = 96MB][act0: 32MB][act1: 32MB]
//   [WiP: 6*192*1024 = 1.125MB][WhP: 6*96*1024 = 0.5625MB]
// ---------------------------------------------------------------------------
extern "C" void kernel_launch(void* const* d_in, const int* in_sizes, int n_in,
                              void* d_out, int out_size, void* d_ws, size_t ws_size,
                              hipStream_t stream) {
    (void)in_sizes; (void)n_in; (void)out_size; (void)ws_size;
    const float* x  = (const float*)d_in[0];
    const float* Wi = (const float*)d_in[1];
    const float* Wh = (const float*)d_in[2];
    const float* bh = (const float*)d_in[3];
    float* out = (float*)d_out;

    char* ws = (char*)d_ws;
    const size_t XG_BYTES  = (size_t)2 * BTN * H3 * 2;     // 96 MB
    const size_t ACT_BYTES = (size_t)BTN * DIMX * 2;       // 32 MB
    const size_t WIP_BYTES = (size_t)6 * WI_FRAGS * FRAG_ELEMS * 2;
    __bf16* xg   = (__bf16*)ws;
    __bf16* act0 = (__bf16*)(ws + XG_BYTES);
    __bf16* act1 = (__bf16*)(ws + XG_BYTES + ACT_BYTES);
    __bf16* WiP  = (__bf16*)(ws + XG_BYTES + 2 * ACT_BYTES);
    __bf16* WhP  = (__bf16*)(ws + XG_BYTES + 2 * ACT_BYTES + WIP_BYTES);

    cvt_f32_bf16_kernel<<<(BTN * DIMX + 255) / 256, 256, 0, stream>>>(
        x, act0, BTN * DIMX);
    prep_weights_kernel<<<dim3((WI_FRAGS + 7) / 8, 6), 256, 0, stream>>>(
        Wi, WiP, DIMX);
    prep_weights_kernel<<<dim3((WH_FRAGS + 7) / 8, 6), 256, 0, stream>>>(
        Wh, WhP, HH);

    const __bf16* ins[3]  = {act0, act1, act0};
    __bf16*       outs[3] = {act1, act0, nullptr};
    for (int l = 0; l < 3; ++l) {
        xg_gemm_kernel<<<dim3(BTN / 128, H3 / 64, 2), 256, 0, stream>>>(
            ins[l], WiP + (size_t)l * 2 * WI_FRAGS * FRAG_ELEMS, xg);
        gru_scan_kernel<<<2, 384, 0, stream>>>(
            xg, WhP + (size_t)l * 2 * WH_FRAGS * FRAG_ELEMS,
            bh + (size_t)l * 2 * H3, outs[l], out, (l == 2) ? 1 : 0);
    }
}
